// Attention_26792005992859
// MI455X (gfx1250) — matrix operand order repaired and checked by reading, not compiled, not run on hardware
//
#include <hip/hip_runtime.h>
#include <hip/hip_bf16.h>
#include <math.h>

// ---- problem constants (match reference) ----
#define B_      16
#define H_      16
#define D_      128
#define BLK_    16
#define BPS_    128
#define S_      2048
#define SCALE_  0.08838834764831845f

// flash-decoding split: 64 splits per sequence, 32 tokens per split,
// one wave32 per workgroup -> 1024 independent KV streams (bandwidth bound).
#define NSPLIT  64
#define TOKS    (S_ / NSPLIT)          // 32
#define REC_    (32 + H_ * D_)         // per-(b,split) ws record: m[16], l[16], O[16*128]

typedef __attribute__((ext_vector_type(16))) __bf16 v16bf;
typedef __attribute__((ext_vector_type(8)))  float  v8f;
typedef __attribute__((ext_vector_type(4)))  float  v4f;
typedef __attribute__((ext_vector_type(4)))  int    v4i;

#define HAS_ASYNC_LDS __has_builtin(__builtin_amdgcn_global_load_async_to_lds_b128)

#if HAS_ASYNC_LDS
// builtin signature (from hipcc diagnostic): (v4i addrspace(1)*, v4i addrspace(3)*, imm int, imm int)
typedef __attribute__((address_space(1))) v4i* g_v4i_ptr;
typedef __attribute__((address_space(3))) v4i* l_v4i_ptr;
#endif

// ---------------------------------------------------------------------------
// Kernel 1: per-(seq, split) partial attention. One wave (32 lanes).
//   QK^T via v_wmma_f32_16x16x32_bf16 "diagonal trick":
//     D[h,h'] = sum_d Q[h,d] * K[t,h',d];  diag(D) = scores of all 16 heads
//     for token t (K = 128 -> 4 chained WMMAs).
//   V is double-buffered in LDS via global_load_async_to_lds_b128 one token
//   ahead (ASYNCcnt pipeline, zero VGPR cost); PV accumulate is VALU fma.
// ---------------------------------------------------------------------------
__global__ __launch_bounds__(32)
void attn_partial(const float* __restrict__ q,
                  const float* __restrict__ knew,
                  const float* __restrict__ vnew,
                  const float* __restrict__ kc,
                  const float* __restrict__ vc,
                  const int*   __restrict__ btab,
                  const int*   __restrict__ smap,
                  const int*   __restrict__ clen,
                  float*       __restrict__ ws)
{
#if HAS_ASYNC_LDS
    // 2 x 8KB V staging buffers; chunk k occupies 512B, lane-interleaved 16B.
    __shared__ float vbuf[2][2048];
#endif

    const int wg    = blockIdx.x;
    const int b     = wg / NSPLIT;
    const int split = wg % NSPLIT;
    const int lane  = threadIdx.x;
    const int m16   = lane & 15;     // A-row M / B-col N / owned head h
    const int hi    = lane >> 4;     // K-dim half selector per ISA layout

    const int ctx = clen[b];
    const int t0  = split * TOKS;

    // slot_mapping is tiny & uniform: keep in registers
    int sm[B_];
    #pragma unroll
    for (int i = 0; i < B_; ++i) sm[i] = smap[i];

    // V row pointer for token t (with functional new-token substitution)
    auto vrow_of = [&](int t) -> const float* {
        const int blk  = btab[b * BPS_ + (t >> 4)];
        const int slot = blk * BLK_ + (t & 15);
        int sub = -1;
        #pragma unroll
        for (int i = 0; i < B_; ++i) if (sm[i] == slot) sub = i;
        return ((sub >= 0) ? (vnew + (size_t)(sub * H_ + m16) * D_)
                           : (vc   + ((size_t)slot * H_ + m16) * D_)) + hi * 64;
    };

#if HAS_ASYNC_LDS
    // stage token tt2's V tile (256B/lane = 16 async b128 ops/wave)
    auto stage_v = [&](int tt2) {
        const float* src = vrow_of(t0 + tt2);
        float* dst = &vbuf[tt2 & 1][lane * 4];
        #pragma unroll
        for (int k = 0; k < 16; ++k)
            __builtin_amdgcn_global_load_async_to_lds_b128(
                (g_v4i_ptr)(src + k * 4), (l_v4i_ptr)(dst + k * 128), 0, 0);
    };
#endif

    // ---- Q -> bf16 A fragments (16x32 each, 4 chunks cover D=128) ----
    // 16-bit A 16x32 layout: lane m16 row M; element e = K(hi*16+e),
    // within the 32-wide chunk (matches B fragment order).
    v16bf afrag[4];
    {
        const float* qrow = q + (size_t)(b * H_ + m16) * D_;
        #pragma unroll
        for (int c = 0; c < 4; ++c) {
            const float* p0 = qrow + c * 32 + hi * 16;
            #pragma unroll
            for (int e = 0; e < 8; ++e) afrag[c][e]     = (__bf16)p0[e];
            #pragma unroll
            for (int e = 0; e < 8; ++e) afrag[c][8 + e] = (__bf16)p0[8 + e];
        }
    }

    // online-softmax state for head m16 (lanes hi=0/1 hold identical copies)
    float m_run = -INFINITY;
    float l_run = 0.0f;
    float oacc[64];
    #pragma unroll
    for (int j = 0; j < 64; ++j) oacc[j] = 0.0f;

#if HAS_ASYNC_LDS
    if (t0 < ctx) stage_v(0);
#endif

    for (int tt = 0; tt < TOKS; ++tt) {
        const int t = t0 + tt;
        if (t >= ctx) break;                         // uniform branch

        const int blk  = btab[b * BPS_ + (t >> 4)];
        const int slot = blk * BLK_ + (t & 15);
        int sub = -1;
        #pragma unroll
        for (int i = 0; i < B_; ++i) if (sm[i] == slot) sub = i;

        // cache layout: flat index = (slot*H + h)*D + d
        const float* krow = (sub >= 0)
            ? (knew + (size_t)(sub * H_ + m16) * D_)
            : (kc   + ((size_t)slot * H_ + m16) * D_);

#if HAS_ASYNC_LDS
        const bool have_next = (tt + 1 < TOKS) && (t + 1 < ctx);
        if (have_next) stage_v(tt + 1);              // overlap with QK WMMAs
#else
        const float* vrow = vrow_of(t);
        __builtin_prefetch(vrow, 0, 0);              // global_prefetch_b8
#endif

        // ---- scores for all 16 heads via WMMA chain over d ----
        // 16-bit B 32x16 fragment, contiguous k as in A: lane col N=m16; element e = K(hi*16+e).
        // K stream has zero reuse vs 192MB L2 -> non-temporal loads.
        v8f acc = {0.f,0.f,0.f,0.f,0.f,0.f,0.f,0.f};
        #pragma unroll
        for (int c = 0; c < 4; ++c) {
            const float* kp = krow + c * 32 + hi * 16;
            v16bf bfrag;
            #pragma unroll
            for (int k4 = 0; k4 < 4; ++k4) {
                v4f kv = __builtin_nontemporal_load((const v4f*)(kp + k4 * 4));
                bfrag[k4 * 4 + 0] = (__bf16)kv[0];
                bfrag[k4 * 4 + 1] = (__bf16)kv[1];
                bfrag[k4 * 4 + 2] = (__bf16)kv[2];
                bfrag[k4 * 4 + 3] = (__bf16)kv[3];
            }
            acc = __builtin_amdgcn_wmma_f32_16x16x32_bf16(
                false, afrag[c], false, bfrag, (short)0, acc, false, false);
        }

        // ---- extract diag(D): head h lives at lane h (h<8, vgpr h) or
        // lane h+16 (h>=8, vgpr h-8). ridx = lane&7 selects the vgpr. ----
        const int ridx = lane & 7;
        float diag = 0.0f;
        #pragma unroll
        for (int r = 0; r < 8; ++r) diag = (ridx == r) ? acc[r] : diag;
        const int srcl = (m16 < 8) ? m16 : (m16 + 16);
        const float s = __shfl(diag, srcl, 32) * SCALE_;

        // ---- online softmax for head h = m16 ----
        const float m_new = fmaxf(m_run, s);
        const float alpha = __expf(m_run - m_new);   // m_run=-inf -> 0
        const float p     = __expf(s - m_new);
        l_run = l_run * alpha + p;
        m_run = m_new;

        // ---- PV accumulate (64 f32 per lane) ----
#if HAS_ASYNC_LDS
        // wait for this token's V copies; next token's 16 may stay in flight
        // (async loads complete in order -> cnt<=16 implies ours are done)
        if (have_next) asm volatile("s_wait_asynccnt 0x10" ::: "memory");
        else           asm volatile("s_wait_asynccnt 0x0"  ::: "memory");
        const float* vsrc = &vbuf[tt & 1][lane * 4];
        #pragma unroll
        for (int k = 0; k < 16; ++k) {
            v4f vv = *(const v4f*)(vsrc + k * 128);  // ds_load_b128
            oacc[4 * k + 0] = oacc[4 * k + 0] * alpha + p * vv[0];
            oacc[4 * k + 1] = oacc[4 * k + 1] * alpha + p * vv[1];
            oacc[4 * k + 2] = oacc[4 * k + 2] * alpha + p * vv[2];
            oacc[4 * k + 3] = oacc[4 * k + 3] * alpha + p * vv[3];
        }
#else
        #pragma unroll
        for (int j = 0; j < 64; j += 4) {
            v4f vv = __builtin_nontemporal_load((const v4f*)(vrow + j));
            oacc[j + 0] = oacc[j + 0] * alpha + p * vv[0];
            oacc[j + 1] = oacc[j + 1] * alpha + p * vv[1];
            oacc[j + 2] = oacc[j + 2] * alpha + p * vv[2];
            oacc[j + 3] = oacc[j + 3] * alpha + p * vv[3];
        }
#endif
    }

    // ---- write partial record: [m(16), l(16), O(16x128)] ----
    float* wsb = ws + (size_t)(b * NSPLIT + split) * REC_;
    if (hi == 0) {
        wsb[m16]      = m_run;
        wsb[16 + m16] = l_run;
    }
    float* op = wsb + 32 + m16 * D_ + hi * 64;
    #pragma unroll
    for (int j = 0; j < 64; j += 4) {
        v4f o4 = { oacc[j], oacc[j + 1], oacc[j + 2], oacc[j + 3] };
        *(v4f*)(op + j) = o4;
    }
}

// ---------------------------------------------------------------------------
// Kernel 2: merge split partials with streaming log-sum-exp.
// ---------------------------------------------------------------------------
__global__ __launch_bounds__(256)
void attn_reduce(const float* __restrict__ ws, float* __restrict__ out)
{
    const int b = blockIdx.x;
    const float* base = ws + (size_t)b * NSPLIT * REC_;
    for (int idx = threadIdx.x; idx < H_ * D_; idx += 256) {
        const int h = idx >> 7;
        float M = -INFINITY, L = 0.0f, O = 0.0f;
        for (int i = 0; i < NSPLIT; ++i) {
            const float* r = base + (size_t)i * REC_;
            const float mi = r[h];
            if (mi == -INFINITY) continue;           // fully-masked split
            const float li = r[16 + h];
            const float oi = r[32 + idx];
            const float Mn = fmaxf(M, mi);
            const float a  = __expf(M - Mn);
            const float g  = __expf(mi - Mn);
            L = L * a + li * g;
            O = O * a + oi * g;
            M = Mn;
        }
        out[(size_t)b * (H_ * D_) + idx] = (L > 0.0f) ? (O / L) : 0.0f;
    }
}

// ---------------------------------------------------------------------------
extern "C" void kernel_launch(void* const* d_in, const int* in_sizes, int n_in,
                              void* d_out, int out_size, void* d_ws, size_t ws_size,
                              hipStream_t stream)
{
    (void)in_sizes; (void)n_in; (void)out_size; (void)ws_size;
    const float* q    = (const float*)d_in[0];
    const float* knew = (const float*)d_in[1];
    const float* vnew = (const float*)d_in[2];
    const float* kc   = (const float*)d_in[3];
    const float* vc   = (const float*)d_in[4];
    const int*   bt   = (const int*)d_in[5];
    const int*   smap = (const int*)d_in[6];
    const int*   cl   = (const int*)d_in[7];
    float*       ws   = (float*)d_ws;    // needs B*NSPLIT*REC_*4 = ~8.5 MB

    attn_partial<<<dim3(B_ * NSPLIT), dim3(32), 0, stream>>>(
        q, knew, vnew, kc, vc, bt, smap, cl, ws);
    attn_reduce<<<dim3(B_), dim3(256), 0, stream>>>(ws, (float*)d_out);
}
